// ODE_RNN_86139864089265
// MI455X (gfx1250) — matrix operand order repaired and checked by reading, not compiled, not run on hardware
//
#include <hip/hip_runtime.h>
#include <math.h>

// ---------------- types ----------------
typedef __attribute__((ext_vector_type(16))) __bf16 v16bf;
typedef __attribute__((ext_vector_type(8)))  __bf16 v8bf;
typedef __attribute__((ext_vector_type(8)))  float  v8f;

#define LSEQ  50
#define BATCH 256
#define DIN   256
#define LAT   512
#define HODE  1024
#define HID   1024     // 2*LAT
#define XLD   257      // data row = 256 features + dt

// ---------------- fragment loaders (layouts per CDNA5 ISA 7.12.2) ----------------
// A-frag (16x32 bf16): lane L holds row m=L&15; K-halves: lanes 0-15 get K=[0..7],[16..23],
// lanes 16-31 get K=[8..15],[24..31]. Caller passes p = &src[m*ld + k0 + (lane>>4)*8].
__device__ __forceinline__ v16bf make_a_from_f32(const float* __restrict__ p) {
  v16bf a;
#pragma unroll
  for (int j = 0; j < 8; ++j) {
    a[j]     = (__bf16)p[j];        // K = kh*8 + j
    a[j + 8] = (__bf16)p[j + 16];   // K = 16 + kh*8 + j
  }
  return a;
}
__device__ __forceinline__ v16bf load_a_bf16(const __bf16* __restrict__ p) {
  v8bf lo = *(const v8bf*)p;
  v8bf hi = *(const v8bf*)(p + 16);
  v16bf a;
#pragma unroll
  for (int j = 0; j < 8; ++j) { a[j] = lo[j]; a[j + 8] = hi[j]; }
  return a;
}
// B-frag (32x16 bf16): lane L holds column n=L&15, K = (L>>4)*8 + [0..7], 16 + (L>>4)*8 + [0..7].
// Weights stored "N-major": element (k,n) at w[n*ldk + k] -> two contiguous 8-bf16 runs per lane (k, k+16).
__device__ __forceinline__ v16bf load_b(const __bf16* __restrict__ w, int ldk, int n, int k) {
  const __bf16* p = w + (size_t)n * ldk + k;
  v8bf lo = *(const v8bf*)p;
  v8bf hi = *(const v8bf*)(p + 16);
  v16bf b;
#pragma unroll
  for (int j = 0; j < 8; ++j) { b[j] = lo[j]; b[j + 8] = hi[j]; }
  return b;
}
__device__ __forceinline__ v8f wmma_bf16(v16bf a, v16bf b, v8f c) {
  return __builtin_amdgcn_wmma_f32_16x16x32_bf16(false, a, false, b, (short)0, c, false, false);
}
__device__ __forceinline__ float sigmoidf(float x) { return 1.0f / (1.0f + expf(-x)); }

#define ZERO8 {0.f,0.f,0.f,0.f,0.f,0.f,0.f,0.f}

// ---------------- one-time conversion / init kernels ----------------
// mlp_w1 (LAT x HODE, K-major) -> w1t (HODE x LAT, N-major bf16)
__global__ void k_cvt_w1t(const float* __restrict__ w1, __bf16* __restrict__ w1t) {
  int idx = blockIdx.x * blockDim.x + threadIdx.x;
  if (idx >= HODE * LAT) return;
  int n = idx / LAT, k = idx % LAT;
  w1t[idx] = (__bf16)w1[(size_t)k * HODE + n];
}
// mlp_w2 (HODE x LAT) -> w2t (LAT x HODE, N-major bf16)
__global__ void k_cvt_w2t(const float* __restrict__ w2, __bf16* __restrict__ w2t) {
  int idx = blockIdx.x * blockDim.x + threadIdx.x;
  if (idx >= LAT * HODE) return;
  int n = idx / HODE, k = idx % HODE;
  w2t[idx] = (__bf16)w2[(size_t)k * LAT + n];
}
// gru_wih/gru_whh are already (3*HID x K) i.e. N-major: straight convert
__global__ void k_cvt(const float* __restrict__ s, __bf16* __restrict__ d, int n) {
  int i = blockIdx.x * blockDim.x + threadIdx.x;
  if (i < n) d[i] = (__bf16)s[i];
}
__global__ void k_zero(float* __restrict__ p, int n) {
  int i = blockIdx.x * blockDim.x + threadIdx.x;
  if (i < n) p[i] = 0.0f;
}

// ---------------- ODE GEMM 1: H = tanh(Y @ W1 + b1) ----------------
// grid (HODE/64, BATCH/64), block 128 (4 waves); wave tile 16x64 (4 WMMA n-subtiles)
__global__ __launch_bounds__(128) void k_mlp1(const float* __restrict__ ySrc,
                                              const __bf16* __restrict__ w1t,
                                              const float* __restrict__ b1,
                                              __bf16* __restrict__ Hbf) {
  int lane = threadIdx.x & 31, wave = threadIdx.x >> 5;
  int m0 = blockIdx.y * 64 + wave * 16;
  int n0 = blockIdx.x * 64;
  int mA  = m0 + (lane & 15);
  int khA = (lane >> 4) * 8;
  int nB  = lane & 15;
  int kB  = (lane >> 4) * 8;
  v8f acc0 = ZERO8, acc1 = ZERO8, acc2 = ZERO8, acc3 = ZERO8;
  for (int k0 = 0; k0 < LAT; k0 += 32) {
    v16bf a = make_a_from_f32(ySrc + (size_t)mA * LAT + k0 + khA);
    acc0 = wmma_bf16(a, load_b(w1t, LAT, n0 +  0 + nB, k0 + kB), acc0);
    acc1 = wmma_bf16(a, load_b(w1t, LAT, n0 + 16 + nB, k0 + kB), acc1);
    acc2 = wmma_bf16(a, load_b(w1t, LAT, n0 + 32 + nB, k0 + kB), acc2);
    acc3 = wmma_bf16(a, load_b(w1t, LAT, n0 + 48 + nB, k0 + kB), acc3);
  }
  v8f acc[4] = {acc0, acc1, acc2, acc3};
  int mBase = m0 + ((lane >> 4) * 8);
#pragma unroll
  for (int s = 0; s < 4; ++s) {
    int n = n0 + s * 16 + (lane & 15);
    float bias = b1[n];
#pragma unroll
    for (int j = 0; j < 8; ++j) {
      Hbf[(size_t)(mBase + j) * HODE + n] = (__bf16)tanhf(acc[s][j] + bias);
    }
  }
}

// ---------------- ODE GEMM 2: yDst = yBase + h_step*(H @ W2 + b2) ----------------
// grid (LAT/64, BATCH/64), block 128
__global__ __launch_bounds__(128) void k_mlp2(const __bf16* __restrict__ Hbf,
                                              const __bf16* __restrict__ w2t,
                                              const float* __restrict__ b2,
                                              const float* __restrict__ yBase,
                                              float* __restrict__ yDst,
                                              const float* __restrict__ xrow) {
  int lane = threadIdx.x & 31, wave = threadIdx.x >> 5;
  int m0 = blockIdx.y * 64 + wave * 16;
  int n0 = blockIdx.x * 64;
  int mA  = m0 + (lane & 15);
  int khA = (lane >> 4) * 8;
  int nB  = lane & 15;
  int kB  = (lane >> 4) * 8;
  v8f acc0 = ZERO8, acc1 = ZERO8, acc2 = ZERO8, acc3 = ZERO8;
  for (int k0 = 0; k0 < HODE; k0 += 32) {
    v16bf a = load_a_bf16(Hbf + (size_t)mA * HODE + k0 + khA);
    acc0 = wmma_bf16(a, load_b(w2t, HODE, n0 +  0 + nB, k0 + kB), acc0);
    acc1 = wmma_bf16(a, load_b(w2t, HODE, n0 + 16 + nB, k0 + kB), acc1);
    acc2 = wmma_bf16(a, load_b(w2t, HODE, n0 + 32 + nB, k0 + kB), acc2);
    acc3 = wmma_bf16(a, load_b(w2t, HODE, n0 + 48 + nB, k0 + kB), acc3);
  }
  v8f acc[4] = {acc0, acc1, acc2, acc3};
  int mBase = m0 + ((lane >> 4) * 8);
  float hstep[8];
#pragma unroll
  for (int j = 0; j < 8; ++j)
    hstep[j] = xrow[(size_t)(mBase + j) * XLD + 256] * (1.0f / 3.0f); // dt / n_sub
#pragma unroll
  for (int s = 0; s < 4; ++s) {
    int n = n0 + s * 16 + (lane & 15);
    float bias = b2[n];
#pragma unroll
    for (int j = 0; j < 8; ++j) {
      size_t idx = (size_t)(mBase + j) * LAT + n;
      yDst[idx] = yBase[idx] + hstep[j] * (acc[s][j] + bias);
    }
  }
}

// ---------------- fused GRU: gates + state update ----------------
// hcat = [yInt | sPrev]; gi = x@Wih^T, gh = hcat@Whh^T; r/z combined, n-gate parts separate.
// grid (HID/32, BATCH/64), block 128; wave tile 16 rows x 32 gate-cols, 4 accumulator sets.
__global__ __launch_bounds__(128) void k_gru(const float* __restrict__ xrow,
                                             const float* __restrict__ yInt,
                                             const float* __restrict__ sPrev,
                                             const __bf16* __restrict__ wihB,
                                             const float* __restrict__ bih,
                                             const __bf16* __restrict__ whhB,
                                             const float* __restrict__ bhh,
                                             float* __restrict__ sNext,
                                             float* __restrict__ outStates,
                                             float* __restrict__ outFinal) {
  int lane = threadIdx.x & 31, wave = threadIdx.x >> 5;
  int m0 = blockIdx.y * 64 + wave * 16;
  int n0 = blockIdx.x * 32;
  int mA  = m0 + (lane & 15);
  int khA = (lane >> 4) * 8;
  int nB  = lane & 15;
  int kB  = (lane >> 4) * 8;
  v8f ar[2] = {ZERO8, ZERO8}, az[2] = {ZERO8, ZERO8};
  v8f ai[2] = {ZERO8, ZERO8}, ah[2] = {ZERO8, ZERO8};
  // x contribution: K = DIN = 256 (x row inside data, ld = 257)
  for (int k0 = 0; k0 < DIN; k0 += 32) {
    v16bf a = make_a_from_f32(xrow + (size_t)mA * XLD + k0 + khA);
#pragma unroll
    for (int s = 0; s < 2; ++s) {
      int nW = n0 + s * 16 + nB;
      ar[s] = wmma_bf16(a, load_b(wihB, DIN, nW,           k0 + kB), ar[s]);
      az[s] = wmma_bf16(a, load_b(wihB, DIN, nW + HID,     k0 + kB), az[s]);
      ai[s] = wmma_bf16(a, load_b(wihB, DIN, nW + 2 * HID, k0 + kB), ai[s]);
    }
  }
  // hcat contribution: K = HID = 1024; each 8-chunk lies fully in one half (512 % 8 == 0)
  for (int k0 = 0; k0 < HID; k0 += 32) {
    const float* src = (k0 < LAT)
        ? (yInt  + (size_t)mA * LAT + k0 + khA)
        : (sPrev + (size_t)mA * LAT + (k0 - LAT) + khA);
    v16bf a = make_a_from_f32(src);
#pragma unroll
    for (int s = 0; s < 2; ++s) {
      int nW = n0 + s * 16 + nB;
      ar[s] = wmma_bf16(a, load_b(whhB, HID, nW,           k0 + kB), ar[s]);
      az[s] = wmma_bf16(a, load_b(whhB, HID, nW + HID,     k0 + kB), az[s]);
      ah[s] = wmma_bf16(a, load_b(whhB, HID, nW + 2 * HID, k0 + kB), ah[s]);
    }
  }
  int mBase = m0 + ((lane >> 4) * 8);
#pragma unroll
  for (int s = 0; s < 2; ++s) {
    int n = n0 + s * 16 + (lane & 15);
    float b_r = bih[n] + bhh[n];
    float b_z = bih[HID + n] + bhh[HID + n];
    float b_i = bih[2 * HID + n];
    float b_h = bhh[2 * HID + n];
    bool lo = (n < LAT);
#pragma unroll
    for (int j = 0; j < 8; ++j) {
      int m = mBase + j;
      float r  = sigmoidf(ar[s][j] + b_r);
      float z  = sigmoidf(az[s][j] + b_z);
      float nn = tanhf(ai[s][j] + b_i + r * (ah[s][j] + b_h));
      float hv = lo ? yInt[(size_t)m * LAT + n] : sPrev[(size_t)m * LAT + (n - LAT)];
      float nh = (1.0f - z) * nn + z * hv;
      outFinal[(size_t)m * HID + n] = nh;
      if (lo) {
        outStates[(size_t)m * LAT + n] = nh;
        sNext[(size_t)m * LAT + n]     = nh;
      }
    }
  }
}

// ---------------- host launcher ----------------
extern "C" void kernel_launch(void* const* d_in, const int* in_sizes, int n_in,
                              void* d_out, int out_size, void* d_ws, size_t ws_size,
                              hipStream_t stream) {
  const float* data = (const float*)d_in[0];
  const float* w1   = (const float*)d_in[1];
  const float* b1   = (const float*)d_in[2];
  const float* w2   = (const float*)d_in[3];
  const float* b2   = (const float*)d_in[4];
  const float* wih  = (const float*)d_in[5];
  const float* bih  = (const float*)d_in[6];
  const float* whh  = (const float*)d_in[7];
  const float* bhh  = (const float*)d_in[8];

  char* p = (char*)d_ws;
  __bf16* w1t  = (__bf16*)p; p += (size_t)HODE * LAT * 2;
  __bf16* w2t  = (__bf16*)p; p += (size_t)LAT * HODE * 2;
  __bf16* wihB = (__bf16*)p; p += (size_t)3 * HID * DIN * 2;
  __bf16* whhB = (__bf16*)p; p += (size_t)3 * HID * HID * 2;
  __bf16* Hbf  = (__bf16*)p; p += (size_t)BATCH * HODE * 2;
  float*  sA   = (float*)p;  p += (size_t)BATCH * LAT * 4;
  float*  sB   = (float*)p;  p += (size_t)BATCH * LAT * 4;
  float*  yW   = (float*)p;  p += (size_t)BATCH * LAT * 4;

  const int thr = 256;
  k_cvt_w1t<<<(HODE * LAT + thr - 1) / thr, thr, 0, stream>>>(w1, w1t);
  k_cvt_w2t<<<(LAT * HODE + thr - 1) / thr, thr, 0, stream>>>(w2, w2t);
  k_cvt<<<(3 * HID * DIN + thr - 1) / thr, thr, 0, stream>>>(wih, wihB, 3 * HID * DIN);
  k_cvt<<<(3 * HID * HID + thr - 1) / thr, thr, 0, stream>>>(whh, whhB, 3 * HID * HID);
  k_zero<<<(BATCH * LAT + thr - 1) / thr, thr, 0, stream>>>(sA, BATCH * LAT);

  float* outStatesBase = (float*)d_out;
  float* outFinal = outStatesBase + (size_t)LSEQ * BATCH * LAT;

  dim3 g1(HODE / 64, BATCH / 64);
  dim3 g2(LAT / 64, BATCH / 64);
  dim3 gg(HID / 32, BATCH / 64);

  for (int t = 0; t < LSEQ; ++t) {
    float* sPrev = (t & 1) ? sB : sA;
    float* sNext = (t & 1) ? sA : sB;
    const float* xrow = data + (size_t)t * BATCH * XLD;
    const float* ySrc = sPrev;
    for (int i = 0; i < 3; ++i) {   // n_sub = 3 Euler sub-steps
      k_mlp1<<<g1, 128, 0, stream>>>(ySrc, w1t, b1, Hbf);
      k_mlp2<<<g2, 128, 0, stream>>>(Hbf, w2t, b2, ySrc, yW, xrow);
      ySrc = yW;
    }
    k_gru<<<gg, 128, 0, stream>>>(xrow, yW, sPrev, wihB, bih, whhB, bhh,
                                  sNext, outStatesBase + (size_t)t * BATCH * LAT, outFinal);
  }
}
